// Net_23553600651528
// MI455X (gfx1250) — compile-verified, matrix operand order checked by reading, not run on hardware
//
#include <hip/hip_runtime.h>
#include <hip/hip_bf16.h>

// ---------------------------------------------------------------------------
// Types / helpers
// ---------------------------------------------------------------------------
typedef unsigned short u16;
typedef __attribute__((ext_vector_type(16))) __bf16 v16bf;
typedef __attribute__((ext_vector_type(8)))  __bf16 v8bf;
typedef __attribute__((ext_vector_type(8)))  float  v8f;

static __device__ __forceinline__ u16 f2bf(float f) {
    union { float f; unsigned int u; } v; v.f = f;
    unsigned int r = v.u + 0x7FFFu + ((v.u >> 16) & 1u);   // RNE
    return (u16)(r >> 16);
}
static __device__ __forceinline__ float bf2f(u16 h) {
    union { unsigned int u; float f; } v; v.u = ((unsigned int)h) << 16; return v.f;
}
// Build a 16x(bf16) WMMA fragment from two contiguous 8-element chunks
// (elements 0..7 = K=clo..clo+7, elements 8..15 = K=clo+16..clo+23).
static __device__ __forceinline__ v16bf mk16(const u16* p0, const u16* p1) {
    v8bf lo = *(const v8bf*)p0;
    v8bf hi = *(const v8bf*)p1;
    v16bf r;
#pragma unroll
    for (int i = 0; i < 8; ++i) { r[i] = lo[i]; r[i + 8] = hi[i]; }
    return r;
}

// ---------------------------------------------------------------------------
// Problem constants
// ---------------------------------------------------------------------------
#define BATCH   2048
#define C1H     26
#define C1W     26
#define C1C     32
#define C2H     24
#define C2W     24
#define C2C     64
#define FC1K    9216     // 64*12*12
#define FC1N    128
#define NQ      10
#define NSTATE  1024

// conv2 implicit GEMM: M = 2048*24*24 = 1,179,648 ; K = 288 = 9 ksteps of 32
#define C2_M        (BATCH * C2H * C2W)
#define C2_MTILES   (C2_M / 16)            // 73728
#define C2_KSTEPS   9

// fc1 GEMM: M = 2048, N = 128, K = 9216 -> 288 ksteps of 32
#define F1_MTILES   (BATCH / 16)           // 128
#define F1_NTILES   (FC1N / 16)            // 8
#define F1_KSTEPS   (FC1K / 32)            // 288

// ---------------------------------------------------------------------------
// 0a. pack conv2 weights into per-lane WMMA B fragments
//     layout: w2frag[kstep(9)][ntile(4)][lane(32)][elem(16)]  (bf16)
// ---------------------------------------------------------------------------
__global__ __launch_bounds__(256) void pack_w2_kernel(const float* __restrict__ w2,
                                                      u16* __restrict__ dst) {
    int t = blockIdx.x * 256 + threadIdx.x;
    if (t >= C2_KSTEPS * 4 * 32 * 16) return;
    int i    = t & 15;
    int lane = (t >> 4) & 31;
    int nt   = (t >> 9) & 3;
    int ks   = t >> 11;
    int clo  = (lane < 16) ? 0 : 8;
    int k    = clo + ((i < 8) ? i : i + 8);     // channel index 0..31
    int n    = nt * 16 + (lane & 15);           // output channel 0..63
    int ky   = ks / 3, kx = ks - ky * 3;
    // conv2_w is OIHW [64][32][3][3]
    float v = w2[((n * 32 + k) * 3 + ky) * 3 + kx];
    dst[t] = f2bf(v);
}

// ---------------------------------------------------------------------------
// 0b. pack fc1 weights: w1frag[kstep(288)][ntile(8)][lane(32)][elem(16)] (bf16)
// ---------------------------------------------------------------------------
__global__ __launch_bounds__(256) void pack_w1_kernel(const float* __restrict__ w1,
                                                      u16* __restrict__ dst) {
    int t = blockIdx.x * 256 + threadIdx.x;
    if (t >= F1_KSTEPS * F1_NTILES * 32 * 16) return;
    int i    = t & 15;
    int lane = (t >> 4) & 31;
    int nt   = (t >> 9) & 7;
    int ks   = t >> 12;
    int clo  = (lane < 16) ? 0 : 8;
    int k    = ks * 32 + clo + ((i < 8) ? i : i + 8);
    int n    = nt * 16 + (lane & 15);
    dst[t] = f2bf(w1[n * FC1K + k]);            // fc1_w is [128][9216]
}

// ---------------------------------------------------------------------------
// 1. conv1 (direct, fp32 math) + ReLU -> NHWC bf16 [2048][26][26][32]
// ---------------------------------------------------------------------------
__global__ __launch_bounds__(256) void conv1_kernel(const float* __restrict__ x,
                                                    const float* __restrict__ w,
                                                    const float* __restrict__ bias,
                                                    u16* __restrict__ out) {
    int idx = blockIdx.x * 256 + threadIdx.x;
    if (idx >= BATCH * C1H * C1W) return;
    int b = idx / (C1H * C1W);
    int r = idx - b * (C1H * C1W);
    int y = r / C1W;
    int xo = r - y * C1W;
    const float* xin = x + b * 784 + y * 28 + xo;
    float pix[9];
#pragma unroll
    for (int ky = 0; ky < 3; ++ky)
#pragma unroll
        for (int kx = 0; kx < 3; ++kx)
            pix[ky * 3 + kx] = xin[ky * 28 + kx];
    u16* op = out + (size_t)idx * C1C;
    for (int o = 0; o < C1C; ++o) {
        float acc = bias[o];
        const float* wp = w + o * 9;
#pragma unroll
        for (int t = 0; t < 9; ++t) acc = fmaf(pix[t], wp[t], acc);
        acc = acc > 0.f ? acc : 0.f;
        op[o] = f2bf(acc);
    }
}

// ---------------------------------------------------------------------------
// 2. conv2 as implicit GEMM with v_wmma_f32_16x16x32_bf16.
//    Block = 128 threads (4 waves); each wave owns one 16(M)x64(N) tile.
//    Packed weights (36,864 B) are staged in LDS once per block.
//    Epilogue: +bias, ReLU -> NHWC bf16 [2048][24][24][64]
// ---------------------------------------------------------------------------
__global__ __launch_bounds__(128) void conv2_wmma_kernel(const u16* __restrict__ a1,
                                                         const u16* __restrict__ w2frag,
                                                         const float* __restrict__ bias,
                                                         u16* __restrict__ out) {
    __shared__ __align__(32) u16 bs[C2_KSTEPS * 4 * 32 * 16];   // 36,864 bytes
    int tid = threadIdx.x;
    { // cooperative stage of all B fragments
        const uint4* src = (const uint4*)w2frag;
        uint4*       dsv = (uint4*)bs;
        for (int i = tid; i < (C2_KSTEPS * 4 * 32 * 16) / 8; i += 128) dsv[i] = src[i];
    }
    __syncthreads();

    int wave = tid >> 5, lane = tid & 31;
    int mtile = blockIdx.x * 4 + wave;
    int clo = (lane < 16) ? 0 : 8;

    // A row assigned to this lane for fragment loading
    int arow = mtile * 16 + (lane & 15);
    int b = arow / (C2H * C2W);
    int rem = arow - b * (C2H * C2W);
    int y = rem / C2W;
    int x = rem - y * C2W;
    const u16* abase = a1 + (size_t)((b * C1H + y) * C1W + x) * C1C + clo;

    v8f acc[4];
#pragma unroll
    for (int nt = 0; nt < 4; ++nt)
#pragma unroll
        for (int r = 0; r < 8; ++r) acc[nt][r] = 0.f;

#pragma unroll
    for (int ks = 0; ks < C2_KSTEPS; ++ks) {
        int ky = ks / 3, kx = ks - ky * 3;
        const u16* ap = abase + (ky * C1W + kx) * C1C;
        v16bf A = mk16(ap, ap + 16);
#pragma unroll
        for (int nt = 0; nt < 4; ++nt) {
            const u16* bp = &bs[((ks * 4 + nt) * 32 + lane) * 16];
            v16bf B = mk16(bp, bp + 8);
            acc[nt] = __builtin_amdgcn_wmma_f32_16x16x32_bf16(
                false, A, false, B, (short)0, acc[nt], false, false);
        }
    }

    // Epilogue: D layout -> row = r + 8*(lane>=16), col = (lane&15) + 16*nt
    int nlo = lane & 15;
    int rbase = mtile * 16 + 8 * (lane >> 4);
#pragma unroll
    for (int r = 0; r < 8; ++r) {
        int p = rbase + r;
        int bb = p / (C2H * C2W);
        int rr = p - bb * (C2H * C2W);
        int yy = rr / C2W;
        int xx = rr - yy * C2W;
        u16* op = out + (size_t)((bb * C2H + yy) * C2W + xx) * C2C + nlo;
#pragma unroll
        for (int nt = 0; nt < 4; ++nt) {
            float v = acc[nt][r] + bias[nt * 16 + nlo];
            v = v > 0.f ? v : 0.f;
            op[nt * 16] = f2bf(v);
        }
    }
}

// ---------------------------------------------------------------------------
// 3. 2x2 maxpool + repack to fc1's flat order: A1[b][c*144 + y*12 + x] (bf16)
// ---------------------------------------------------------------------------
__global__ __launch_bounds__(256) void pool_kernel(const u16* __restrict__ c2,
                                                   u16* __restrict__ a1) {
    int t = blockIdx.x * 256 + threadIdx.x;
    if (t >= BATCH * FC1K) return;
    int b = t / FC1K;
    int r = t - b * FC1K;
    int c = r / 144;
    int s = r - c * 144;
    int y = s / 12, x = s - y * 12;
    const u16* base = c2 + (size_t)((b * C2H + 2 * y) * C2W + 2 * x) * C2C + c;
    u16 e0 = base[0], e1 = base[C2C], e2 = base[C2W * C2C], e3 = base[C2W * C2C + C2C];
    u16 m = e0;
    if (bf2f(e1) > bf2f(m)) m = e1;
    if (bf2f(e2) > bf2f(m)) m = e2;
    if (bf2f(e3) > bf2f(m)) m = e3;
    a1[t] = m;
}

// ---------------------------------------------------------------------------
// 4. fc1 GEMM with WMMA bf16. 1024 waves; wave = (mtile, ntile).
//    Epilogue: +bias, ReLU -> bf16 h[2048][128]
// ---------------------------------------------------------------------------
__global__ __launch_bounds__(128) void fc1_wmma_kernel(const u16* __restrict__ a1,
                                                       const u16* __restrict__ w1frag,
                                                       const float* __restrict__ bias,
                                                       u16* __restrict__ h) {
    int tid = threadIdx.x;
    int wave = tid >> 5, lane = tid & 31;
    int wid = blockIdx.x * 4 + wave;         // 0..1023
    int mtile = wid >> 3;
    int nt = wid & 7;
    int clo = (lane < 16) ? 0 : 8;

    int arow = mtile * 16 + (lane & 15);
    const u16* abase = a1 + (size_t)arow * FC1K + clo;

    v8f acc;
#pragma unroll
    for (int r = 0; r < 8; ++r) acc[r] = 0.f;

    for (int ks = 0; ks < F1_KSTEPS; ++ks) {
        const u16* ap = abase + ks * 32;
        __builtin_prefetch(ap + 8 * 32, 0, 0);         // global_prefetch on A stream
        v16bf A = mk16(ap, ap + 16);
        const u16* bp = w1frag + ((size_t)(ks * F1_NTILES + nt) * 32 + lane) * 16;
        v16bf B = mk16(bp, bp + 8);
        acc = __builtin_amdgcn_wmma_f32_16x16x32_bf16(
            false, A, false, B, (short)0, acc, false, false);
    }

    int n = nt * 16 + (lane & 15);
    float bia = bias[n];
    int rbase = mtile * 16 + 8 * (lane >> 4);
#pragma unroll
    for (int r = 0; r < 8; ++r) {
        float v = acc[r] + bia;
        v = v > 0.f ? v : 0.f;
        h[(size_t)(rbase + r) * FC1N + n] = f2bf(v);
    }
}

// ---------------------------------------------------------------------------
// 5. fc2 + sigmoid*2pi -> angles [2048][10] (fp32)
// ---------------------------------------------------------------------------
__global__ __launch_bounds__(256) void fc2_kernel(const u16* __restrict__ h,
                                                  const float* __restrict__ w,
                                                  const float* __restrict__ bias,
                                                  float* __restrict__ angles) {
    int t = blockIdx.x * 256 + threadIdx.x;
    if (t >= BATCH * NQ) return;
    int b = t / NQ, j = t - b * NQ;
    float acc = bias[j];
    const u16* hp = h + (size_t)b * FC1N;
    const float* wp = w + j * FC1N;
    for (int k = 0; k < FC1N; ++k) acc = fmaf(bf2f(hp[k]), wp[k], acc);
    float sig = 1.f / (1.f + expf(-acc));
    angles[t] = 6.283185307179586f * sig;
}

// ---------------------------------------------------------------------------
// 6. 10-qubit state-vector simulation, one block per sample, state in LDS.
//    Wire w maps to bit (9 - w) of the flat index (reference axis order).
// ---------------------------------------------------------------------------
#define F2c(a, b) make_float2((a), (b))
static __device__ __forceinline__ float2 cmul(float2 a, float2 b) {
    return make_float2(a.x * b.x - a.y * b.y, a.x * b.y + a.y * b.x);
}
static __device__ __forceinline__ float2 cadd(float2 a, float2 b) {
    return make_float2(a.x + b.x, a.y + b.y);
}

static __device__ void apply1(float2* st, int w, float2 u00, float2 u01,
                              float2 u10, float2 u11, int tid) {
    int sh = 9 - w;
    int stride = 1 << sh;
    for (int q = tid; q < 512; q += 256) {
        int i0 = ((q >> sh) << (sh + 1)) | (q & (stride - 1));
        int i1 = i0 | stride;
        float2 p = st[i0], r = st[i1];
        st[i0] = cadd(cmul(u00, p), cmul(u01, r));
        st[i1] = cadd(cmul(u10, p), cmul(u11, r));
    }
    __syncthreads();
}
static __device__ void rx_g(float2* st, int w, float t, int tid) {
    float c = cosf(0.5f * t), s = sinf(0.5f * t);
    apply1(st, w, F2c(c, 0), F2c(0, -s), F2c(0, -s), F2c(c, 0), tid);
}
static __device__ void ry_g(float2* st, int w, float t, int tid) {
    float c = cosf(0.5f * t), s = sinf(0.5f * t);
    apply1(st, w, F2c(c, 0), F2c(-s, 0), F2c(s, 0), F2c(c, 0), tid);
}
static __device__ void rz_g(float2* st, int w, float t, int tid) {
    float c = cosf(0.5f * t), s = sinf(0.5f * t);
    apply1(st, w, F2c(c, -s), F2c(0, 0), F2c(0, 0), F2c(c, s), tid);
}
static __device__ void phase_g(float2* st, int w, float t, int tid) {
    apply1(st, w, F2c(1, 0), F2c(0, 0), F2c(0, 0), F2c(cosf(t), sinf(t)), tid);
}
static __device__ void rot_g(float2* st, int w, float phi, float th, float om, int tid) {
    float c = cosf(0.5f * th), s = sinf(0.5f * th);
    float ap = 0.5f * (phi + om), am = 0.5f * (phi - om);
    float2 m00 = F2c(cosf(ap) * c, -sinf(ap) * c);   //  e^{-i ap} c
    float2 m01 = F2c(-cosf(am) * s, -sinf(am) * s);  // -e^{ i am} s
    float2 m10 = F2c(cosf(am) * s, -sinf(am) * s);   //  e^{-i am} s
    float2 m11 = F2c(cosf(ap) * c, sinf(ap) * c);    //  e^{ i ap} c
    apply1(st, w, m00, m01, m10, m11, tid);
}
static __device__ void cz_g(float2* st, int w0, int w1, int tid) {
    int m0 = 1 << (9 - w0), m1 = 1 << (9 - w1);
    for (int i = tid; i < NSTATE; i += 256)
        if ((i & m0) && (i & m1)) { st[i].x = -st[i].x; st[i].y = -st[i].y; }
    __syncthreads();
}
static __device__ void cx_g(float2* st, int wc, int wt, int tid) {
    int mc = 1 << (9 - wc), mt = 1 << (9 - wt);
    for (int i = tid; i < NSTATE; i += 256)
        if ((i & mc) && !(i & mt)) { float2 t0 = st[i]; st[i] = st[i | mt]; st[i | mt] = t0; }
    __syncthreads();
}
static __device__ void cy_g(float2* st, int wc, int wt, int tid) {
    int mc = 1 << (9 - wc), mt = 1 << (9 - wt);
    for (int i = tid; i < NSTATE; i += 256)
        if ((i & mc) && !(i & mt)) {
            float2 a0 = st[i], a1 = st[i | mt];
            st[i]      = F2c(a1.y, -a1.x);   // -i * a1
            st[i | mt] = F2c(-a0.y, a0.x);   //  i * a0
        }
    __syncthreads();
}
static __device__ void swap_g(float2* st, int w0, int w1, int tid) {
    int m0 = 1 << (9 - w0), m1 = 1 << (9 - w1);
    for (int i = tid; i < NSTATE; i += 256)
        if ((i & m0) && !(i & m1)) {
            int j = i ^ m0 ^ m1;
            float2 t0 = st[i]; st[i] = st[j]; st[j] = t0;
        }
    __syncthreads();
}
static __device__ void cswap_g(float2* st, int wc, int wa, int wb, int tid) {
    int mc = 1 << (9 - wc), ma = 1 << (9 - wa), mb = 1 << (9 - wb);
    for (int i = tid; i < NSTATE; i += 256)
        if ((i & mc) && (i & ma) && !(i & mb)) {
            int j = i ^ ma ^ mb;
            float2 t0 = st[i]; st[i] = st[j]; st[j] = t0;
        }
    __syncthreads();
}
static __device__ void toff_g(float2* st, int wc0, int wc1, int wt, int tid) {
    int m0 = 1 << (9 - wc0), m1 = 1 << (9 - wc1), mt = 1 << (9 - wt);
    for (int i = tid; i < NSTATE; i += 256)
        if ((i & m0) && (i & m1) && !(i & mt)) {
            float2 t0 = st[i]; st[i] = st[i | mt]; st[i | mt] = t0;
        }
    __syncthreads();
}

__global__ __launch_bounds__(256) void qsim_kernel(const float* __restrict__ angles,
                                                   const float* __restrict__ theta0,
                                                   const float* __restrict__ theta_rz,
                                                   const float* __restrict__ theta_ps,
                                                   const float* __restrict__ rot_p,
                                                   float* __restrict__ out) {
    __shared__ float2 st[NSTATE];
    __shared__ float red[256];
    __shared__ float ev[NQ];
    int b = blockIdx.x, tid = threadIdx.x;

    float a[NQ];
#pragma unroll
    for (int j = 0; j < NQ; ++j) a[j] = angles[b * NQ + j];
    float trz = theta_rz[0], tps = theta_ps[0];

    for (int i = tid; i < NSTATE; i += 256)
        st[i] = make_float2(i == 0 ? 1.f : 0.f, 0.f);
    __syncthreads();

    for (int k = 0; k < NQ; ++k) rx_g(st, k, theta0[k], tid);   // TrainableRxAll
    for (int k = 0; k < NQ; ++k) rx_g(st, k, a[k], tid);        // RxAll(angles)
    rx_g(st, 1, a[1], tid);
    rx_g(st, 5, -0.7853981633974483f, tid);                     // RX(-pi/4)
    ry_g(st, 2, a[2], tid);
    rz_g(st, 3, a[3], tid);
    apply1(st, 4, F2c(1, 0), F2c(0, 0), F2c(0, 0), F2c(0, 1), tid);                 // S
    apply1(st, 5, F2c(1, 0), F2c(0, 0), F2c(0, 0),
           F2c(0.70710678118f, 0.70710678118f), tid);                               // T
    rz_g(st, 6, trz, tid);
    apply1(st, 7, F2c(0.5f, 0.5f), F2c(0.5f, -0.5f),
           F2c(0.5f, -0.5f), F2c(0.5f, 0.5f), tid);                                 // SX
    cz_g(st, 0, 5, tid);
    cx_g(st, 0, 5, tid);
    cy_g(st, 0, 5, tid);
    cy_g(st, 3, 8, tid);
    swap_g(st, 2, 3, tid);
    cswap_g(st, 4, 5, 6, tid);
    toff_g(st, 8, 5, 0, tid);
    phase_g(st, 8, tps, tid);
    phase_g(st, 7, a[7], tid);
    rot_g(st, 4, rot_p[0], rot_p[1], rot_p[2], tid);
    rot_g(st, 5, a[6], a[7], a[8], tid);

    // <Y_k> = 2 * sum_pairs Im(conj(psi_i0) * psi_i1)
    for (int k = 0; k < NQ; ++k) {
        int sh = 9 - k, stride = 1 << sh;
        float s = 0.f;
        for (int q = tid; q < 512; q += 256) {
            int i0 = ((q >> sh) << (sh + 1)) | (q & (stride - 1));
            float2 p = st[i0], r = st[i0 | stride];
            s += 2.f * (p.x * r.y - p.y * r.x);
        }
        red[tid] = s;
        __syncthreads();
        for (int o = 128; o > 0; o >>= 1) {
            if (tid < o) red[tid] += red[tid + o];
            __syncthreads();
        }
        if (tid == 0) ev[k] = red[0];
        __syncthreads();
    }
    if (tid == 0) {
        float m = ev[0];
        for (int k = 1; k < NQ; ++k) m = fmaxf(m, ev[k]);
        float se = 0.f;
        for (int k = 0; k < NQ; ++k) se += expf(ev[k] - m);
        float lse = m + logf(se);
        for (int k = 0; k < NQ; ++k) out[b * NQ + k] = ev[k] - lse;
    }
}

// ---------------------------------------------------------------------------
// Launcher
// ---------------------------------------------------------------------------
extern "C" void kernel_launch(void* const* d_in, const int* in_sizes, int n_in,
                              void* d_out, int out_size, void* d_ws, size_t ws_size,
                              hipStream_t stream) {
    const float* x        = (const float*)d_in[0];
    const float* conv1_w  = (const float*)d_in[1];
    const float* conv1_b  = (const float*)d_in[2];
    const float* conv2_w  = (const float*)d_in[3];
    const float* conv2_b  = (const float*)d_in[4];
    const float* fc1_w    = (const float*)d_in[5];
    const float* fc1_b    = (const float*)d_in[6];
    const float* fc2_w    = (const float*)d_in[7];
    const float* fc2_b    = (const float*)d_in[8];
    const float* theta0   = (const float*)d_in[9];
    const float* theta_rz = (const float*)d_in[10];
    const float* theta_ps = (const float*)d_in[11];
    const float* rot_p    = (const float*)d_in[12];
    float* out = (float*)d_out;

    char* ws = (char*)d_ws;
    // region 1 (conv1out, later reused for A1 / h / angles once conv1out is dead)
    const size_t SZ_C1  = (size_t)BATCH * C1H * C1W * C1C * 2;       // 88,604,672
    const size_t OFF_C1 = 0;
    const size_t OFF_A1 = 0;                                          // reuse after conv2
    const size_t OFF_H  = OFF_A1 + (size_t)BATCH * FC1K * 2;          // 37,748,736
    const size_t OFF_AN = OFF_H + (size_t)BATCH * FC1N * 2;           // +524,288
    // region 2 (conv2out)
    const size_t OFF_C2  = SZ_C1;
    const size_t SZ_C2   = (size_t)BATCH * C2H * C2W * C2C * 2;       // 150,994,944
    // region 3 (packed weights)
    const size_t OFF_W2F = OFF_C2 + SZ_C2;
    const size_t SZ_W2F  = (size_t)C2_KSTEPS * 4 * 32 * 16 * 2;       // 36,864
    const size_t OFF_W1F = OFF_W2F + SZ_W2F;
    const size_t SZ_W1F  = (size_t)F1_KSTEPS * F1_NTILES * 32 * 16 * 2; // 2,359,296
    if (ws_size < OFF_W1F + SZ_W1F) return;                           // need ~231 MB

    u16*   c1out  = (u16*)(ws + OFF_C1);
    u16*   c2out  = (u16*)(ws + OFF_C2);
    u16*   a1buf  = (u16*)(ws + OFF_A1);
    u16*   hbuf   = (u16*)(ws + OFF_H);
    float* angles = (float*)(ws + OFF_AN);
    u16*   w2frag = (u16*)(ws + OFF_W2F);
    u16*   w1frag = (u16*)(ws + OFF_W1F);

    // 0. pack weights into WMMA fragment order
    pack_w2_kernel<<<(C2_KSTEPS * 4 * 32 * 16 + 255) / 256, 256, 0, stream>>>(conv2_w, w2frag);
    pack_w1_kernel<<<(F1_KSTEPS * F1_NTILES * 32 * 16 + 255) / 256, 256, 0, stream>>>(fc1_w, w1frag);

    // 1. conv1 -> NHWC bf16
    conv1_kernel<<<(BATCH * C1H * C1W + 255) / 256, 256, 0, stream>>>(x, conv1_w, conv1_b, c1out);

    // 2. conv2 implicit GEMM (WMMA bf16), 4 waves / block
    conv2_wmma_kernel<<<C2_MTILES / 4, 128, 0, stream>>>(c1out, w2frag, conv2_b, c2out);

    // 3. maxpool + repack to fc1 K-order (overwrites conv1out region)
    pool_kernel<<<(BATCH * FC1K + 255) / 256, 256, 0, stream>>>(c2out, a1buf);

    // 4. fc1 (WMMA bf16)
    fc1_wmma_kernel<<<(F1_MTILES * F1_NTILES) / 4, 128, 0, stream>>>(a1buf, w1frag, fc1_b, hbuf);

    // 5. fc2 + sigmoid -> angles
    fc2_kernel<<<(BATCH * NQ + 255) / 256, 256, 0, stream>>>(hbuf, fc2_w, fc2_b, angles);

    // 6. quantum circuit + <Y_k> + log_softmax -> d_out
    qsim_kernel<<<BATCH, 256, 0, stream>>>(angles, theta0, theta_rz, theta_ps, rot_p, out);
}
